// NodePhases_20985210208767
// MI455X (gfx1250) — 8-bit matrix operand order repaired and checked by reading, compiled, not run on hardware
//
#include <hip/hip_runtime.h>

typedef int v8i __attribute__((ext_vector_type(8)));

#define T_DIM 16
#define P_DIM 64
#define NT_PER_WAVE 4

__device__ __forceinline__ int pack01(int4 x) {
    unsigned lo = __builtin_amdgcn_perm((unsigned)x.y, (unsigned)x.x, 0x0C0C0400u);
    unsigned hi = __builtin_amdgcn_perm((unsigned)x.w, (unsigned)x.z, 0x0C0C0400u);
    return (int)__builtin_amdgcn_perm(hi, lo, 0x05040100u);
}


__global__ __launch_bounds__(256) void pauli_prod_kernel(
    const int* __restrict__ phases,
    const int* __restrict__ params,
    const int* __restrict__ counts,
    const int* __restrict__ pvals,
    float* __restrict__ out,
    int G, int B)
{
    const int lane = threadIdx.x & 31;
    const int wave = threadIdx.x >> 5;
    const int gid  = blockIdx.x * 8 + wave;
    const int g    = gid >> 2;
    const int grp  = gid & 3;
    if (g >= G) return;

    const int half = lane >> 4;
    const int mrow = lane & 15;

    v8i A;
    {
        const int* abase = params + ((size_t)g * T_DIM + mrow) * P_DIM;
        #pragma unroll
        for (int v = 0; v < 8; ++v) {
            const int kbase = 16 * (v >> 1) + 8 * half + 4 * (v & 1);
            A[v] = pack01(*reinterpret_cast<const int4*>(abase + kbase));
        }
    }

    const int4 pl  = *reinterpret_cast<const int4*>(phases + (size_t)g * T_DIM + 8 * half);
    const int4 ph2 = *reinterpret_cast<const int4*>(phases + (size_t)g * T_DIM + 8 * half + 4);
    const int ph[8] = { pl.x, pl.y, pl.z, pl.w, ph2.x, ph2.y, ph2.z, ph2.w };
    const int cnt = counts[g];

    bool jb0[8], jb1[8];
    int  f0[8], nf0[8];
    #pragma unroll
    for (int d = 0; d < 8; ++d) {
        const int p = ph[d] & 7;
        jb0[d] = (p & 1) != 0;
        jb1[d] = (p & 2) != 0;
        const int s0  = (p <= 2 || p == 7) ? 1 : -1;
        const bool act = (d + 8 * half) < cnt;
        f0[d]  = act ? s0 : 0;
        nf0[d] = -f0[d];
    }

    #pragma unroll
    for (int ni = 0; ni < NT_PER_WAVE; ++ni) {
        const int nt = grp * NT_PER_WAVE + ni;

        v8i Bm;
        {
            const int* bbase = pvals + ((size_t)(nt * 16 + mrow)) * P_DIM;
            #pragma unroll
            for (int v = 0; v < 8; ++v) {
                const int kbase = 16 * (v >> 1) + 8 * half + 4 * (v & 1);
                Bm[v] = pack01(*reinterpret_cast<const int4*>(bbase + kbase));
            }
        }

        v8i C = {};
        v8i D = __builtin_amdgcn_wmma_i32_16x16x64_iu8(false, A, false, Bm, C, false, false);

        int r0, r1, r2, r3;
        {
            const bool par = (D[0] & 1) != 0;
            const int f = par ? nf0[0] : f0[0];
            r0 = 1 + ((!jb0[0] && !jb1[0]) ? f : 0);
            r1 = (( jb0[0] && !jb1[0]) ? f : 0);
            r2 = ((!jb0[0] &&  jb1[0]) ? f : 0);
            r3 = (( jb0[0] &&  jb1[0]) ? f : 0);
        }
        #pragma unroll
        for (int d = 1; d < 8; ++d) {
            const bool par = (D[d] & 1) != 0;
            const int f  = par ? nf0[d] : f0[d];
            const int n1 = -r1, n2 = -r2, n3 = -r3;
            const int v0 = jb1[d] ? (jb0[d] ? r1 : n2) : (jb0[d] ? r3 : r0);
            const int v1 = jb1[d] ? (jb0[d] ? r2 : r3) : (jb0[d] ? r0 : r1);
            const int v2 = jb1[d] ? (jb0[d] ? n3 : r0) : (jb0[d] ? r1 : r2);
            const int v3 = jb1[d] ? (jb0[d] ? r0 : n1) : (jb0[d] ? n2 : r3);
            r0 += f * v0;
            r1 += f * v1;
            r2 += f * v2;
            r3 += f * v3;
        }

        const int y0 = __shfl_xor(r0, 16, 32);
        const int y1 = __shfl_xor(r1, 16, 32);
        const int y2 = __shfl_xor(r2, 16, 32);
        const int y3 = __shfl_xor(r3, 16, 32);
        const int z0 = r0 * y0 + r1 * y3 + r3 * y1 - r2 * y2;
        const int z1 = r0 * y1 + r1 * y0 + r2 * y3 + r3 * y2;
        const int z2 = r0 * y2 + r1 * y1 + r2 * y0 - r3 * y3;
        const int z3 = r0 * y3 - r1 * y2 - r2 * y1 + r3 * y0;

        if (half == 0) {
            const int b = nt * 16 + mrow;
            float4 res = make_float4((float)z0, (float)z1, (float)z2, (float)z3);
            *reinterpret_cast<float4*>(out + ((size_t)b * G + g) * 4) = res;
        }
    }
}

extern "C" void kernel_launch(void* const* d_in, const int* in_sizes, int n_in,
                              void* d_out, int out_size, void* d_ws, size_t ws_size,
                              hipStream_t stream) {
    const int* phases = (const int*)d_in[0];
    const int* params = (const int*)d_in[1];
    const int* counts = (const int*)d_in[2];
    const int* pvals  = (const int*)d_in[3];

    const int G = in_sizes[2];
    const int P = in_sizes[1] / in_sizes[0];
    const int B = in_sizes[3] / P;
    float* out = (float*)d_out;

    const long long waves = (long long)G * 4;
    const int blocks = (int)((waves + 7) / 8);
    pauli_prod_kernel<<<blocks, 256, 0, stream>>>(phases, params, counts, pvals, out, G, B);
}
